// MultiHeadSelfAttention_28647431865041
// MI455X (gfx1250) — compile-verified, matrix operand order checked by reading, not run on hardware
//
#include <hip/hip_runtime.h>
#include <hip/hip_bf16.h>

typedef __attribute__((ext_vector_type(16))) __bf16 v16bf;
typedef __attribute__((ext_vector_type(8)))  float  v8f;

#define EMBED  768
#define QKV3   2304
#define HEADS  12
#define HD     64
#define SEQ    1024
#define BATCH  8
#define MTOT   (BATCH*SEQ)   // 8192
#define QSCALE 0.180336879f  // Hd^-0.5 * log2(e) folded into Q

// ---------- helpers ----------

__device__ __forceinline__ unsigned short f2bf(float x) {
  unsigned int u = __float_as_uint(x);
  return (unsigned short)((u + 0x7FFFu + ((u >> 16) & 1u)) >> 16);  // RNE
}

__device__ __forceinline__ float exp2_fast(float x) {
  return __builtin_amdgcn_exp2f(x);   // v_exp_f32
}

__device__ __forceinline__ v8f wmma_bf16(v16bf a, v16bf b, v8f c) {
  return __builtin_amdgcn_wmma_f32_16x16x32_bf16(false, a, false, b, (short)0, c,
                                                 false, false);
}

// Async global->LDS copy of 16 bytes (ASYNCcnt-tracked, no VGPR round trip).
__device__ __forceinline__ void async_b128(void* lds, const void* glob) {
  unsigned l = (unsigned)(uintptr_t)lds;
  unsigned long long g = (unsigned long long)(uintptr_t)glob;
  asm volatile("global_load_async_to_lds_b128 %0, %1, off"
               :: "v"(l), "v"(g) : "memory");
}
// Wait until ASYNCcnt <= N (async completions are in-order per wave).
template <int N>
__device__ __forceinline__ void wait_async() {
  asm volatile("s_wait_asynccnt %0" :: "i"(N) : "memory");
}

union FragU { uint4 q[2]; v16bf v; };

// 16x32 A-frag / 32x16 B-frag loader per ISA 7.12.2 (16-bit layouts):
// lane L holds row/col L&15; two contiguous 16B reads at K = 8*(L>>4), 16+8*(L>>4).
__device__ __forceinline__ v16bf frag_ld(const unsigned short* p, int row0,
                                         int col0, int stride) {
  int lane = threadIdx.x & 31;
  int r = row0 + (lane & 15);
  int c = col0 + ((lane >> 4) << 3);
  FragU u;
  u.q[0] = *(const uint4*)(p + r * stride + c);
  u.q[1] = *(const uint4*)(p + r * stride + c + 16);
  return u.v;
}

// ---------- fp32 -> bf16 conversion ----------

__global__ __launch_bounds__(256) void cvt_bf16_kernel(const float* __restrict__ src,
                                                       unsigned short* __restrict__ dst,
                                                       int n) {
  int i = blockIdx.x * blockDim.x + threadIdx.x;
  int stride = gridDim.x * blockDim.x;
  for (; i < n; i += stride) dst[i] = f2bf(src[i]);
}

// fp32 (K x N, row-major) -> bf16 transposed (N x K, row-major), LDS-tiled.
__global__ __launch_bounds__(256) void cvt_transpose_kernel(
    const float* __restrict__ src, unsigned short* __restrict__ dst, int K, int N) {
  __shared__ float tile[32][33];
  const int k0 = blockIdx.x * 32, n0 = blockIdx.y * 32;
  const int tx = threadIdx.x & 31, ty = threadIdx.x >> 5;  // ty 0..7
  #pragma unroll
  for (int i = 0; i < 4; ++i) {
    int r = ty + i * 8;
    tile[r][tx] = src[(size_t)(k0 + r) * N + n0 + tx];
  }
  __syncthreads();
  #pragma unroll
  for (int i = 0; i < 4; ++i) {
    int r = ty + i * 8;
    dst[(size_t)(n0 + r) * K + k0 + tx] = f2bf(tile[tx][r]);
  }
}

// ---------- tiled bf16 WMMA GEMM ----------
// C(8192 x Ncols) = A(8192 x K) * Wt^T + bias, Wt pre-transposed (N x K).
// Block tile 128x128, BK=32, 256 threads (8 waves, 4x2); wave: 32 rows x 64 cols.
// Triple-buffered LDS tiles, async global->LDS, partial asynccnt waits.
// mode 0: QKV epilogue -> bf16 Q(*scale),K (B,H,N,Hd) and transposed V (B,H,Hd,N)
// mode 1: proj epilogue -> fp32 d_out
__global__ __launch_bounds__(256) void gemm_bf16_kernel(
    const unsigned short* __restrict__ A, const unsigned short* __restrict__ Wt,
    const float* __restrict__ bias, int K, int mode,
    unsigned short* __restrict__ Qo, unsigned short* __restrict__ Ko,
    unsigned short* __restrict__ Vto, float* __restrict__ Co) {
  __shared__ alignas(16) unsigned short At[3][128 * 40];  // 128 rows x 32 k
  __shared__ alignas(16) unsigned short Bt[3][128 * 40];  // 128 n   x 32 k

  const int tid  = threadIdx.x;
  const int lane = tid & 31;
  const int w    = tid >> 5;      // wave 0..7
  const int wm   = w & 3;         // row wave
  const int wn   = w >> 2;        // col wave
  const int grp  = lane >> 4;
  const int ln   = lane & 15;
  const int m0   = blockIdx.x * 128;
  const int n0   = blockIdx.y * 128;

  auto issue = [&](int k0, int buf) {
    #pragma unroll
    for (int i = 0; i < 2; ++i) {              // A tile: 128x32 = 512 b128
      int idx = tid + i * 256;
      int row = idx >> 2, c8 = (idx & 3) << 3;
      async_b128(&At[buf][row * 40 + c8], &A[(size_t)(m0 + row) * K + k0 + c8]);
    }
    #pragma unroll
    for (int i = 0; i < 2; ++i) {              // B tile: 128x32 = 512 b128
      int idx = tid + i * 256;
      int row = idx >> 2, c8 = (idx & 3) << 3;
      async_b128(&Bt[buf][row * 40 + c8], &Wt[(size_t)(n0 + row) * K + k0 + c8]);
    }
  };  // 4 async b128 per thread per tile

  v8f acc[2][4] = {};
  const int steps = K / 32;       // 24
  issue(0, 0);
  issue(32, 1);
  wait_async<4>();                // batch0 landed, batch1 in flight
  __syncthreads();

  for (int s = 0; s < steps; ++s) {
    const int cur = s % 3;
    if (s + 2 < steps) issue((s + 2) * 32, (s + 2) % 3);

    v16bf bfr[4];
    #pragma unroll
    for (int cb = 0; cb < 4; ++cb)
      bfr[cb] = frag_ld(Bt[cur], wn * 64 + cb * 16, 0, 40);
    #pragma unroll
    for (int sub = 0; sub < 2; ++sub) {
      v16bf af = frag_ld(At[cur], wm * 32 + sub * 16, 0, 40);
      #pragma unroll
      for (int cb = 0; cb < 4; ++cb)
        acc[sub][cb] = wmma_bf16(af, bfr[cb], acc[sub][cb]);
    }
    if (s + 2 < steps) wait_async<4>();  // tile s+1 done, s+2 still in flight
    else               wait_async<0>();
    __syncthreads();
  }

  // epilogue
  const int nb  = n0 + wn * 64;          // wave-uniform column base
  const int sdx = nb / EMBED;            // 0=Q 1=K 2=V (mode 0)
  const int h   = (nb % EMBED) / HD;
  #pragma unroll
  for (int sub = 0; sub < 2; ++sub) {
    #pragma unroll
    for (int cb = 0; cb < 4; ++cb) {
      float bv = bias[nb + cb * 16 + ln];
      #pragma unroll
      for (int r = 0; r < 8; ++r) {
        int m = m0 + wm * 32 + sub * 16 + r + 8 * grp;
        float v = acc[sub][cb][r] + bv;
        if (mode == 1) {
          Co[(size_t)m * EMBED + nb + cb * 16 + ln] = v;
        } else {
          int b = m >> 10, npos = m & 1023;
          int d = cb * 16 + ln;
          size_t bhh = (size_t)(b * HEADS + h);
          if (sdx == 0)      Qo[(bhh * SEQ + npos) * HD + d] = f2bf(v * QSCALE);
          else if (sdx == 1) Ko[(bhh * SEQ + npos) * HD + d] = f2bf(v);
          else               Vto[(bhh * HD + d) * SEQ + npos] = f2bf(v);  // V^T
        }
      }
    }
  }
}

// ---------- flash attention ----------
// grid: (N/64, HEADS, BATCH); 128 threads = 4 waves; wave owns 16 query rows.
// Triple-buffered async K / V^T tiles; softmax in exp2 domain (scale pre-folded
// into Q by the QKV GEMM epilogue).
__global__ __launch_bounds__(128) void mhsa_attn_kernel(
    const unsigned short* __restrict__ Q, const unsigned short* __restrict__ Kg,
    const unsigned short* __restrict__ Vt, unsigned short* __restrict__ attn_out) {
  __shared__ alignas(16) unsigned short Kt[3][64 * 72];     // keys x d
  __shared__ alignas(16) unsigned short Vtile[3][64 * 72];  // d x keys
  __shared__ alignas(16) unsigned short Pt[4][16 * 72];     // per-wave P staging

  const int nt   = blockIdx.x;
  const int h    = blockIdx.y;
  const int b    = blockIdx.z;
  const int tid  = threadIdx.x;
  const int lane = tid & 31;
  const int w    = tid >> 5;
  const int grp  = lane >> 4;
  const int ln   = lane & 15;

  const size_t bh = (size_t)(b * HEADS + h);
  const unsigned short* Qbh = Q  + bh * SEQ * HD;
  const unsigned short* Kbh = Kg + bh * SEQ * HD;
  const unsigned short* Vbh = Vt + bh * HD * SEQ;

  auto issue = [&](int key0, int buf) {
    #pragma unroll
    for (int i = 0; i < 4; ++i) {
      int idx = tid + i * 128;                 // 0..511
      int rr = idx >> 3, c8 = (idx & 7) << 3;
      async_b128(&Kt[buf][rr * 72 + c8], &Kbh[(size_t)(key0 + rr) * HD + c8]);
      async_b128(&Vtile[buf][rr * 72 + c8], &Vbh[(size_t)rr * SEQ + key0 + c8]);
    }
  };  // 8 async b128 per thread per chunk

  // Q strip A-fragments straight from global (A layout == 2x16B per lane)
  v16bf qf[2];
  {
    int qrow = nt * 64 + w * 16 + ln;
    #pragma unroll
    for (int kc = 0; kc < 2; ++kc) {
      FragU u;
      u.q[0] = *(const uint4*)&Qbh[(size_t)qrow * HD + kc * 32 + grp * 8];
      u.q[1] = *(const uint4*)&Qbh[(size_t)qrow * HD + kc * 32 + grp * 8 + 16];
      qf[kc] = u.v;
    }
  }

  v8f o[4] = {};
  float rm[8], rl[8];
  #pragma unroll
  for (int r = 0; r < 8; ++r) { rm[r] = -1e30f; rl[r] = 0.f; }

  const int chunks = SEQ / 64;    // 16
  issue(0, 0);
  issue(64, 1);
  wait_async<8>();
  __syncthreads();

  for (int c = 0; c < chunks; ++c) {
    const int cur = c % 3;
    if (c + 2 < chunks) issue((c + 2) * 64, (c + 2) % 3);

    // S = Q K^T   (log2-scaled: Q was pre-multiplied by Hd^-0.5 * log2 e)
    v8f s[4];
    #pragma unroll
    for (int cb = 0; cb < 4; ++cb) {
      v8f sa = {};
      #pragma unroll
      for (int kc = 0; kc < 2; ++kc)
        sa = wmma_bf16(qf[kc], frag_ld(Kt[cur], cb * 16, kc * 32, 72), sa);
      s[cb] = sa;
    }

    // online softmax (base-2); row m = r + 8*grp lives in a 16-lane half-wave
    #pragma unroll
    for (int r = 0; r < 8; ++r) {
      float mx = fmaxf(fmaxf(s[0][r], s[1][r]), fmaxf(s[2][r], s[3][r]));
      #pragma unroll
      for (int off = 8; off >= 1; off >>= 1) mx = fmaxf(mx, __shfl_xor(mx, off, 32));
      float nmax = fmaxf(rm[r], mx);
      float corr = exp2_fast(rm[r] - nmax);
      rm[r] = nmax;
      float rs = 0.f;
      #pragma unroll
      for (int cb = 0; cb < 4; ++cb) {
        float p = exp2_fast(s[cb][r] - nmax);
        s[cb][r] = p;
        rs += p;
      }
      #pragma unroll
      for (int off = 8; off >= 1; off >>= 1) rs += __shfl_xor(rs, off, 32);
      rl[r] = rl[r] * corr + rs;
      #pragma unroll
      for (int cb = 0; cb < 4; ++cb) o[cb][r] *= corr;
    }

    // stage P (C-layout -> A-layout) through wave-private LDS
    #pragma unroll
    for (int cb = 0; cb < 4; ++cb)
      #pragma unroll
      for (int r = 0; r < 8; ++r)
        Pt[w][(r + 8 * grp) * 72 + cb * 16 + ln] = f2bf(s[cb][r]);
    asm volatile("s_wait_dscnt 0x0" ::: "memory");  // same-wave DS is in-order

    // O += P * V
    #pragma unroll
    for (int kc = 0; kc < 2; ++kc) {
      v16bf pf = frag_ld(Pt[w], 0, kc * 32, 72);
      #pragma unroll
      for (int db = 0; db < 4; ++db)
        o[db] = wmma_bf16(pf, frag_ld(Vtile[cur], db * 16, kc * 32, 72), o[db]);
    }

    if (c + 2 < chunks) wait_async<8>();
    else                wait_async<0>();
    __syncthreads();
  }

  // normalize and write bf16 attn output in (B*N, 768) layout for proj GEMM
  #pragma unroll
  for (int db = 0; db < 4; ++db)
    #pragma unroll
    for (int r = 0; r < 8; ++r) {
      int m = nt * 64 + w * 16 + r + 8 * grp;
      float v = o[db][r] / rl[r];
      attn_out[((size_t)b * SEQ + m) * EMBED + h * HD + db * 16 + ln] = f2bf(v);
    }
}

// ---------- launcher ----------

extern "C" void kernel_launch(void* const* d_in, const int* in_sizes, int n_in,
                              void* d_out, int out_size, void* d_ws, size_t ws_size,
                              hipStream_t stream) {
  const float* x      = (const float*)d_in[0];
  const float* W_qkv  = (const float*)d_in[1];
  const float* b_qkv  = (const float*)d_in[2];
  const float* W_proj = (const float*)d_in[3];
  const float* b_proj = (const float*)d_in[4];
  float* out = (float*)d_out;

  char* ws = (char*)d_ws;
  unsigned short* Xbf   = (unsigned short*)(ws);             // 8192x768
  unsigned short* WqkvT = (unsigned short*)(ws + 12582912);  // 2304x768 (transposed)
  unsigned short* WprjT = (unsigned short*)(ws + 16121856);  // 768x768  (transposed)
  unsigned short* Qb    = (unsigned short*)(ws + 17301504);  // (B,H,N,Hd), pre-scaled
  unsigned short* Kb    = (unsigned short*)(ws + 29884416);  // (B,H,N,Hd)
  unsigned short* Vtb   = (unsigned short*)(ws + 42467328);  // (B,H,Hd,N)
  unsigned short* Abf   = (unsigned short*)(ws + 55050240);  // 8192x768

  cvt_bf16_kernel<<<2048, 256, 0, stream>>>(x, Xbf, MTOT * EMBED);
  cvt_transpose_kernel<<<dim3(EMBED / 32, QKV3 / 32), 256, 0, stream>>>(
      W_qkv, WqkvT, EMBED, QKV3);
  cvt_transpose_kernel<<<dim3(EMBED / 32, EMBED / 32), 256, 0, stream>>>(
      W_proj, WprjT, EMBED, EMBED);

  gemm_bf16_kernel<<<dim3(MTOT / 128, QKV3 / 128), 256, 0, stream>>>(
      Xbf, WqkvT, b_qkv, EMBED, /*mode=*/0, Qb, Kb, Vtb, nullptr);

  mhsa_attn_kernel<<<dim3(SEQ / 64, HEADS, BATCH), 128, 0, stream>>>(
      Qb, Kb, Vtb, Abf);

  gemm_bf16_kernel<<<dim3(MTOT / 128, EMBED / 128), 256, 0, stream>>>(
      Abf, WprjT, b_proj, EMBED, /*mode=*/1, nullptr, nullptr, nullptr, out);
}
